// SecondOrderDeformableAlignment_30803505447540
// MI455X (gfx1250) — compile-verified, matrix operand order checked by reading, not run on hardware
//
#include <hip/hip_runtime.h>
#include <hip/hip_bf16.h>
#include <math.h>

// ---------------------------------------------------------------------------
// SecondOrderDeformableAlignment for MI455X (gfx1250), wave32 + WMMA.
// All five stages are implicit-GEMMs on v_wmma_f32_16x16x32_f16.
// Block tile: 64(M) x 64(N), K staged 64-wide in LDS (all K are 64-multiples:
// 1728 / 576 / 1152), 8 waves = 4 M-subtiles x 2 N-groups, 2 accumulators and
// 4 WMMAs per wave per stage.
// ---------------------------------------------------------------------------

typedef __attribute__((ext_vector_type(16))) _Float16 v16h;
typedef __attribute__((ext_vector_type(8)))  _Float16 v8h;
typedef __attribute__((ext_vector_type(8)))  float    v8f;

#define HH     128
#define WW_    128
#define HWSZ   16384            // 128*128
#define BATCH  4
#define NTOT   (BATCH * HWSZ)   // 65536 pixels (GEMM N)
#define NT64   (NTOT / 64)      // 1024 N-tiles of 64
#define KPAD   72               // LDS K row length (64 + pad); 144B row stride

// ---- A-fragment: 16 rows (M) x 32 K of weights, converted f32 -> f16 ------
// ISA layout (16-bit A 16x32): lane = M%16 ; lanes>=16 take K+8 ;
// element i holds K = (i/8)*16 + (lane/16)*8 + (i%8).
__device__ __forceinline__ v16h load_a_frag(const float* __restrict__ Wt,
                                            int row, int M, int Ktot, int kA) {
  v16h a;
  if (row < M) {
    const float* wr = Wt + (size_t)row * (size_t)Ktot + kA;
    float4 q0 = *(const float4*)(wr + 0);
    float4 q1 = *(const float4*)(wr + 4);
    float4 q2 = *(const float4*)(wr + 16);
    float4 q3 = *(const float4*)(wr + 20);
    a[0]  = (_Float16)q0.x; a[1]  = (_Float16)q0.y;
    a[2]  = (_Float16)q0.z; a[3]  = (_Float16)q0.w;
    a[4]  = (_Float16)q1.x; a[5]  = (_Float16)q1.y;
    a[6]  = (_Float16)q1.z; a[7]  = (_Float16)q1.w;
    a[8]  = (_Float16)q2.x; a[9]  = (_Float16)q2.y;
    a[10] = (_Float16)q2.z; a[11] = (_Float16)q2.w;
    a[12] = (_Float16)q3.x; a[13] = (_Float16)q3.y;
    a[14] = (_Float16)q3.z; a[15] = (_Float16)q3.w;
  } else {
#pragma unroll
    for (int j = 0; j < 16; ++j) a[j] = (_Float16)0.f;
  }
  return a;
}

// ---- B-fragment from LDS: lane = N%16, K packing mirrors A ----------------
// bsrow points at one N column's 64 contiguous K halves -> two ds_load_b128.
__device__ __forceinline__ v16h load_b_frag(const _Float16* bsrow, int kOff,
                                            int half) {
  v8h lo = *(const v8h*)(bsrow + kOff + half * 8);
  v8h hi = *(const v8h*)(bsrow + kOff + 16 + half * 8);
  v16h b;
#pragma unroll
  for (int j = 0; j < 8; ++j) { b[j] = lo[j]; b[j + 8] = hi[j]; }
  return b;
}

// ---------------------------------------------------------------------------
// Generic 3x3 conv (pad=1) as implicit GEMM.
//   out[m][n] = sum_k W[m][k] * im2col(In)[k][n],  k = ci*9 + kh*3 + kw
// ACT: 0 = bias only, 1 = leaky relu(0.1), 2 = offset/mask head
//      (m < 288 -> 10*tanh, else sigmoid)
// ---------------------------------------------------------------------------
template <int ACT>
__global__ __launch_bounds__(256)
void conv3x3_wmma_kernel(const float* __restrict__ In,
                         const float* __restrict__ Wt,
                         const float* __restrict__ Bias,
                         float* __restrict__ Out,
                         int Cin, int M) {
  __shared__ __align__(16) _Float16 Bs[64][KPAD];  // 64 N x 64 K

  const int tid    = threadIdx.x;
  const int lane   = tid & 31;
  const int waveId = tid >> 5;
  const int mSub   = waveId & 3;
  const int nGrp   = waveId >> 2;                  // 0 or 1 -> 32 N each
  const int m0     = blockIdx.y * 64 + mSub * 16;
  const int n0     = blockIdx.x * 64;
  const int Ktot   = Cin * 9;                      // multiple of 64
  const int half   = lane >> 4;
  const int col    = lane & 15;
  const int row    = m0 + col;

  v8f acc0 = {0.f, 0.f, 0.f, 0.f, 0.f, 0.f, 0.f, 0.f};
  v8f acc1 = {0.f, 0.f, 0.f, 0.f, 0.f, 0.f, 0.f, 0.f};

  for (int k0 = 0; k0 < Ktot; k0 += 64) {
    // -------- build 64K x 64N im2col tile in LDS (all 256 threads) --------
    for (int e = tid; e < 4096; e += 256) {
      const int kk = e >> 6;
      const int nn = e & 63;
      const int k  = k0 + kk;
      const int ci = k / 9;
      const int kr = k - ci * 9;
      const int kh = kr / 3;
      const int kw = kr - kh * 3;
      float v = 0.f;
      const int n = n0 + nn;
      if (n < NTOT) {
        const int bb = n >> 14;          // / HWSZ
        const int hw = n & 16383;
        const int y  = (hw >> 7) + kh - 1;
        const int x  = (hw & 127) + kw - 1;
        if ((unsigned)y < 128u && (unsigned)x < 128u)
          v = In[((bb * Cin + ci) << 14) + (y << 7) + x];
      }
      Bs[nn][kk] = (_Float16)v;
    }
    __syncthreads();

    // speculative prefetch of next K-stage of weights (global_prefetch_b8)
    __builtin_prefetch(Wt + (size_t)row * (size_t)Ktot + k0 + 64 + half * 8, 0, 0);

    // -------- 2 K-chunks x 2 N-subtiles = 4 WMMAs per wave --------
#pragma unroll
    for (int c = 0; c < 2; ++c) {
      v16h a = load_a_frag(Wt, row, M, Ktot, k0 + c * 32 + half * 8);
      v16h bA = load_b_frag(&Bs[(nGrp << 5) + col][0],      c * 32, half);
      v16h bB = load_b_frag(&Bs[(nGrp << 5) + 16 + col][0], c * 32, half);
      acc0 = __builtin_amdgcn_wmma_f32_16x16x32_f16(false, a, false, bA,
                                                    (short)0, acc0, false, false);
      acc1 = __builtin_amdgcn_wmma_f32_16x16x32_f16(false, a, false, bB,
                                                    (short)0, acc1, false, false);
    }
    __syncthreads();
  }

  // -------- epilogue: bias + activation, store fp32 --------
  const int mAdd = (lane >> 4) << 3;       // C/D layout: M = v + 8*(lane/16)
#pragma unroll
  for (int t = 0; t < 2; ++t) {
    const int n = n0 + (nGrp << 5) + (t << 4) + col;
    const v8f& acc = t ? acc1 : acc0;
    if (n < NTOT) {
      const int bb = n >> 14;
      const int hw = n & 16383;
#pragma unroll
      for (int v = 0; v < 8; ++v) {
        const int m = m0 + mAdd + v;
        if (m < M) {
          float r = acc[v] + Bias[m];
          if (ACT == 1) {
            r = (r > 0.f) ? r : 0.1f * r;
          } else if (ACT == 2) {
            r = (m < 288) ? 10.f * tanhf(r) : 1.f / (1.f + expf(-r));
          }
          Out[(bb * M + m) * HWSZ + hw] = r;
        }
      }
    }
  }
}

// ---------------------------------------------------------------------------
// Deformable conv as implicit GEMM: M=64, K = 128*9 = 1152, N = pixels.
// B-tile element (k,n): bilinear sample of x at (h-1+kh+dy, w-1+kw+dx),
// zero-padded, scaled by sigmoid mask. Offsets/mask read from the activated
// conv4 output ("offmask", B x 432 x H x W).
// ---------------------------------------------------------------------------
__global__ __launch_bounds__(256)
void deform_conv_wmma_kernel(const float* __restrict__ X,
                             const float* __restrict__ OffMask,
                             const float* __restrict__ Wt,
                             const float* __restrict__ Bias,
                             float* __restrict__ Out) {
  __shared__ __align__(16) _Float16 Bs[64][KPAD];

  const int tid    = threadIdx.x;
  const int lane   = tid & 31;
  const int waveId = tid >> 5;
  const int mSub   = waveId & 3;
  const int nGrp   = waveId >> 2;
  const int m0     = mSub * 16;            // M = 64 exactly
  const int n0     = blockIdx.x * 64;
  const int Ktot   = 1152;                 // 128 * 9 = 18 * 64
  const int half   = lane >> 4;
  const int col    = lane & 15;
  const int row    = m0 + col;

  v8f acc0 = {0.f, 0.f, 0.f, 0.f, 0.f, 0.f, 0.f, 0.f};
  v8f acc1 = {0.f, 0.f, 0.f, 0.f, 0.f, 0.f, 0.f, 0.f};

  for (int k0 = 0; k0 < Ktot; k0 += 64) {
    for (int e = tid; e < 4096; e += 256) {
      const int kk = e >> 6;
      const int nn = e & 63;
      const int k  = k0 + kk;
      const int ci = k / 9;                // input channel 0..127
      const int kr = k - ci * 9;           // kernel tap 0..8
      const int g  = ci >> 3;              // deform group 0..15
      float v = 0.f;
      const int n = n0 + nn;
      if (n < NTOT) {
        const int bb = n >> 14;
        const int hw = n & 16383;
        const int h  = hw >> 7;
        const int w  = hw & 127;
        const float* om = OffMask + (size_t)bb * (432 * HWSZ) + hw;
        const float dy = om[(g * 18 + kr * 2 + 0) << 14];
        const float dx = om[(g * 18 + kr * 2 + 1) << 14];
        const float mk = om[(288 + g * 9 + kr) << 14];
        const float py = (float)(h - 1 + kr / 3) + dy;
        const float px = (float)(w - 1 + kr % 3) + dx;
        const float y0f = floorf(py), x0f = floorf(px);
        const float ly = py - y0f, lx = px - x0f;
        const int y0 = (int)y0f, x0 = (int)x0f;
        const float* xc = X + ((bb * 128 + ci) << 14);
        const bool yi0 = (unsigned)y0 < 128u,  yi1 = (unsigned)(y0 + 1) < 128u;
        const bool xi0 = (unsigned)x0 < 128u,  xi1 = (unsigned)(x0 + 1) < 128u;
        const float v00 = (yi0 && xi0) ? xc[(y0 << 7) + x0]           : 0.f;
        const float v01 = (yi0 && xi1) ? xc[(y0 << 7) + x0 + 1]       : 0.f;
        const float v10 = (yi1 && xi0) ? xc[((y0 + 1) << 7) + x0]     : 0.f;
        const float v11 = (yi1 && xi1) ? xc[((y0 + 1) << 7) + x0 + 1] : 0.f;
        v = ((1.f - ly) * (1.f - lx) * v00 + (1.f - ly) * lx * v01 +
             ly * (1.f - lx) * v10 + ly * lx * v11) * mk;
      }
      Bs[nn][kk] = (_Float16)v;
    }
    __syncthreads();

    __builtin_prefetch(Wt + (size_t)row * (size_t)Ktot + k0 + 64 + half * 8, 0, 0);

#pragma unroll
    for (int c = 0; c < 2; ++c) {
      v16h a = load_a_frag(Wt, row, 64, Ktot, k0 + c * 32 + half * 8);
      v16h bA = load_b_frag(&Bs[(nGrp << 5) + col][0],      c * 32, half);
      v16h bB = load_b_frag(&Bs[(nGrp << 5) + 16 + col][0], c * 32, half);
      acc0 = __builtin_amdgcn_wmma_f32_16x16x32_f16(false, a, false, bA,
                                                    (short)0, acc0, false, false);
      acc1 = __builtin_amdgcn_wmma_f32_16x16x32_f16(false, a, false, bB,
                                                    (short)0, acc1, false, false);
    }
    __syncthreads();
  }

  const int mAdd = (lane >> 4) << 3;
#pragma unroll
  for (int t = 0; t < 2; ++t) {
    const int n = n0 + (nGrp << 5) + (t << 4) + col;
    const v8f& acc = t ? acc1 : acc0;
    if (n < NTOT) {
      const int bb = n >> 14;
      const int hw = n & 16383;
#pragma unroll
      for (int v = 0; v < 8; ++v) {
        const int m = m0 + mAdd + v;          // < 64 always
        Out[(bb * 64 + m) * HWSZ + hw] = acc[v] + Bias[m];
      }
    }
  }
}

// ---------------------------------------------------------------------------
// Host-side launcher. Workspace layout (fp32):
//   [0)                offmask : 4*432*16384 floats = 113,246,208 B
//   [113246208)        tA      : 4*64*16384 floats  =  16,777,216 B
//   [130023424)        tB      : 4*64*16384 floats  =  16,777,216 B
// ---------------------------------------------------------------------------
extern "C" void kernel_launch(void* const* d_in, const int* in_sizes, int n_in,
                              void* d_out, int out_size, void* d_ws, size_t ws_size,
                              hipStream_t stream) {
  const float* x     = (const float*)d_in[0];
  const float* extra = (const float*)d_in[1];
  const float* w1    = (const float*)d_in[2];
  const float* b1    = (const float*)d_in[3];
  const float* w2    = (const float*)d_in[4];
  const float* b2    = (const float*)d_in[5];
  const float* w3    = (const float*)d_in[6];
  const float* b3    = (const float*)d_in[7];
  const float* w4    = (const float*)d_in[8];
  const float* b4    = (const float*)d_in[9];
  const float* wdc   = (const float*)d_in[10];
  const float* bdc   = (const float*)d_in[11];
  float*       out   = (float*)d_out;

  char* ws = (char*)d_ws;
  const size_t OFFMASK_BYTES = (size_t)BATCH * 432 * HWSZ * sizeof(float);
  const size_t T_BYTES       = (size_t)BATCH * 64  * HWSZ * sizeof(float);
  float* offmask = (float*)ws;
  float* tA      = (float*)(ws + OFFMASK_BYTES);
  float* tB      = (float*)(ws + OFFMASK_BYTES + T_BYTES);

  const dim3 blk(256);
  const dim3 gN(NT64, 1);          // M=64  -> 1 M-block of 64
  const dim3 gN4(NT64, 7);         // M=432 -> ceil(432/64)=7 M-blocks

  // conv1: extra_feat (192ch) -> tA (64ch), lrelu
  conv3x3_wmma_kernel<1><<<gN, blk, 0, stream>>>(extra, w1, b1, tA, 192, 64);
  // conv2: tA -> tB, lrelu
  conv3x3_wmma_kernel<1><<<gN, blk, 0, stream>>>(tA, w2, b2, tB, 64, 64);
  // conv3: tB -> tA, lrelu
  conv3x3_wmma_kernel<1><<<gN, blk, 0, stream>>>(tB, w3, b3, tA, 64, 64);
  // conv4: tA -> offmask (432ch), fused 10*tanh / sigmoid head
  conv3x3_wmma_kernel<2><<<gN4, blk, 0, stream>>>(tA, w4, b4, offmask, 64, 432);
  // deformable conv: x + offmask -> out (64ch)
  deform_conv_wmma_kernel<<<gN, blk, 0, stream>>>(x, offmask, wdc, bdc, out);
}
